// BertSelfAttention_47081431499374
// MI455X (gfx1250) — compile-verified, matrix operand order checked by reading, not run on hardware
//
#include <hip/hip_runtime.h>
#include <hip/hip_bf16.h>

// ---------- WMMA / TDM types (CDNA5 gfx1250, wave32) ----------
typedef __attribute__((ext_vector_type(16))) __bf16 v16bf;
typedef __attribute__((ext_vector_type(8)))  float  v8f;
typedef __attribute__((ext_vector_type(4)))  unsigned int v4u;
typedef __attribute__((ext_vector_type(8)))  int v8i;
typedef __attribute__((ext_vector_type(4)))  int v4i;

union Frag { v16bf v; unsigned int u[8]; };

// RNE float -> bf16
static __device__ __forceinline__ unsigned short f2bf(float x) {
  unsigned int u = __float_as_uint(x);
  u = (u + 0x7FFFu + ((u >> 16) & 1u)) >> 16;
  return (unsigned short)u;
}
static __device__ __forceinline__ unsigned int f2bf_pack(float a, float b) {
  return (unsigned int)f2bf(a) | ((unsigned int)f2bf(b) << 16);
}
// wave32 lane broadcast via ds_bpermute (byte index)
static __device__ __forceinline__ float lane_bcast(float v, int srcLane) {
  return __int_as_float(__builtin_amdgcn_ds_bpermute(srcLane << 2, __float_as_int(v)));
}

// A/B fragment K-offset pattern (16-bit A 16x32 layout, CDNA5 ISA 7.12.2):
// lane half 0 holds K pairs {0..7,16..23}, half 1 holds {8..15,24..31}.
#define KOFF(v, half) ((((v) >> 2) << 4) + (half) * 8 + (((v) & 3) << 1))

#define NH 16
#define HD 64

// ---------------- Tensor Data Mover (guarded; fallback = manual staging) ----
#if defined(__AMDGCN__) && __has_builtin(__builtin_amdgcn_tensor_load_to_lds) && \
    __has_builtin(__builtin_amdgcn_s_wait_tensorcnt)
#define USE_TDM 1
#else
#define USE_TDM 0
#endif

#if USE_TDM
// 2-D tile DMA: global (row-major, elem=bf16) -> LDS with per-row DWORD padding.
// D# bitfields per CDNA5 ISA.
static __device__ __forceinline__ void tdm_load_2d(
    unsigned lds_addr, const void* gaddr,
    unsigned tensor_d0, unsigned tensor_d1, unsigned long long stride0,
    unsigned tile_d0, unsigned tile_d1,
    unsigned pad_interval_code, unsigned pad_amount_code)
{
  const unsigned long long ga = (unsigned long long)(size_t)gaddr;
  v4u g0;
  g0.x = 1u;                                           // count=1, user D#
  g0.y = lds_addr;                                     // lds_addr [63:32]
  g0.z = (unsigned)(ga & 0xFFFFFFFFu);                 // global_addr [95:64]
  g0.w = (unsigned)((ga >> 32) & 0x01FFFFFFu) | 0x80000000u;  // ga hi | type=2

  v8i g1;
  g1[0] = (int)((1u << 16) |                           // data_size = 2B
                (1u << 20) |                           // pad_enable
                (pad_interval_code << 22) |
                (pad_amount_code << 25));
  g1[1] = (int)(tensor_d0 << 16);                      // tensor_dim0[15:0] @63:48
  g1[2] = (int)((tensor_d0 >> 16) | (tensor_d1 << 16));
  g1[3] = (int)((tensor_d1 >> 16) | (tile_d0 << 16));  // tile_dim0 @127:112
  g1[4] = (int)(tile_d1 & 0xFFFFu);                    // tile_dim1; tile_dim2=0
  g1[5] = (int)(unsigned)(stride0 & 0xFFFFFFFFu);      // tensor_dim0_stride lo
  g1[6] = (int)(unsigned)((stride0 >> 32) & 0xFFFFu);  // stride hi; dim1_stride=0
  g1[7] = 0;

  v4i gz = {0, 0, 0, 0};
#if __clang_major__ >= 23
  v8i gz8 = {0, 0, 0, 0, 0, 0, 0, 0};
  __builtin_amdgcn_tensor_load_to_lds(g0, g1, gz, gz, gz8, 0);
#else
  __builtin_amdgcn_tensor_load_to_lds(g0, g1, gz, gz, 0);
#endif
}
#endif

// =====================================================================
// Projection GEMM: out[s,o] = X[s,:] . W[o,:] + bias[o]   (W is [O,H] row-major)
// Output bf16:  transposed==0 -> [b,h,s,d]   transposed==1 -> [b,h,d,s]
// Block: 256 thr (8 waves), tile 64(M) x 128(N); each wave 32x32; K step 32.
// =====================================================================
__global__ void __launch_bounds__(256)
proj_kernel(const float* __restrict__ X, const float* __restrict__ W,
            const float* __restrict__ bias, unsigned short* __restrict__ out,
            int S, int transposed)
{
  __shared__ unsigned short lds_x[64 * 34];    // 64 rows x 32 bf16 (pad 34)
  __shared__ unsigned short lds_w[128 * 34];   // 128 rows x 32 bf16

  const int tid  = threadIdx.x;
  const int lane = tid & 31;
  const int wave = tid >> 5;          // 0..7
  const int half = lane >> 4;
  const int l16  = lane & 15;

  const int b     = blockIdx.z;
  const int sbase = blockIdx.x * 64;
  const int obase = blockIdx.y * 128;

  const float* Xb = X + (size_t)b * S * 1024;

  const int mb = (wave & 1) * 32;     // wave M base within 64
  const int nb = (wave >> 1) * 32;    // wave N base within 128

  v8f c00 = {}, c01 = {}, c10 = {}, c11 = {};

  const int xrow = tid >> 2, xcol = (tid & 3) * 8;   // X tile loader
  const int wrow = tid >> 1, wcol = (tid & 1) * 16;  // W tile loader

  for (int kb = 0; kb < 1024; kb += 32) {
    __syncthreads();
    {
      const float4* src = (const float4*)(Xb + (size_t)(sbase + xrow) * 1024 + kb + xcol);
      float4 p0 = src[0], p1 = src[1];
      unsigned int* dst = (unsigned int*)lds_x + xrow * 17 + (xcol >> 1);
      dst[0] = f2bf_pack(p0.x, p0.y); dst[1] = f2bf_pack(p0.z, p0.w);
      dst[2] = f2bf_pack(p1.x, p1.y); dst[3] = f2bf_pack(p1.z, p1.w);
    }
    {
      const float4* src = (const float4*)(W + (size_t)(obase + wrow) * 1024 + kb + wcol);
      float4 p0 = src[0], p1 = src[1], p2 = src[2], p3 = src[3];
      unsigned int* dst = (unsigned int*)lds_w + wrow * 17 + (wcol >> 1);
      dst[0] = f2bf_pack(p0.x, p0.y); dst[1] = f2bf_pack(p0.z, p0.w);
      dst[2] = f2bf_pack(p1.x, p1.y); dst[3] = f2bf_pack(p1.z, p1.w);
      dst[4] = f2bf_pack(p2.x, p2.y); dst[5] = f2bf_pack(p2.z, p2.w);
      dst[6] = f2bf_pack(p3.x, p3.y); dst[7] = f2bf_pack(p3.z, p3.w);
    }
    __syncthreads();

    Frag a0, a1, b0, b1;
#pragma unroll
    for (int v = 0; v < 8; ++v) {
      const int ko = KOFF(v, half) >> 1;
      a0.u[v] = ((const unsigned int*)lds_x)[(mb + l16)      * 17 + ko];
      a1.u[v] = ((const unsigned int*)lds_x)[(mb + 16 + l16) * 17 + ko];
      b0.u[v] = ((const unsigned int*)lds_w)[(nb + l16)      * 17 + ko];
      b1.u[v] = ((const unsigned int*)lds_w)[(nb + 16 + l16) * 17 + ko];
    }
    c00 = __builtin_amdgcn_wmma_f32_16x16x32_bf16(false, a0.v, false, b0.v, (short)0, c00, false, false);
    c01 = __builtin_amdgcn_wmma_f32_16x16x32_bf16(false, a0.v, false, b1.v, (short)0, c01, false, false);
    c10 = __builtin_amdgcn_wmma_f32_16x16x32_bf16(false, a1.v, false, b0.v, (short)0, c10, false, false);
    c11 = __builtin_amdgcn_wmma_f32_16x16x32_bf16(false, a1.v, false, b1.v, (short)0, c11, false, false);
  }

  // Epilogue: + bias, fp32 -> bf16, scatter to [b,h,s,d] or [b,h,d,s]
#pragma unroll
  for (int j = 0; j < 2; ++j) {
    const int n  = obase + nb + j * 16 + l16;       // 0..1023
    const float bn = bias[n];
    const int hh = n >> 6;
    const int dd = n & 63;
    const size_t bh = (size_t)b * NH + hh;
#pragma unroll
    for (int i = 0; i < 2; ++i) {
      v8f& c = (j == 0) ? (i == 0 ? c00 : c10) : (i == 0 ? c01 : c11);
#pragma unroll
      for (int r = 0; r < 8; ++r) {
        const int s = sbase + mb + i * 16 + r + 8 * half;
        const float val = c[r] + bn;
        const size_t idx = transposed
            ? (bh * HD + dd) * (size_t)S + s
            : (bh * (size_t)S + s) * HD + dd;
        out[idx] = f2bf(val);
      }
    }
  }
}

// =====================================================================
// Fused flash attention over concatenated [self(1024, masked) | cross(512)] keys.
// Grid (S/64, NH, B); block 128 thr = 4 waves; each wave owns 16 queries.
// Q,K bf16 [b,h,s,d]; V bf16 TRANSPOSED [b,h,d,s]; out fp32 [b,s,h*d].
// K/V tiles staged into LDS by the Tensor Data Mover when available.
// =====================================================================
__global__ void __launch_bounds__(128)
attn_kernel(const unsigned short* __restrict__ Q,
            const unsigned short* __restrict__ K,
            const unsigned short* __restrict__ VT,
            const unsigned short* __restrict__ QO,
            const unsigned short* __restrict__ KO,
            const unsigned short* __restrict__ VOT,
            const float* __restrict__ mask,
            float* __restrict__ out)
{
  __shared__ unsigned short lds_kt[32 * 66];       // 32 keys x 64 d (pad 66)
  __shared__ unsigned short lds_vt[64 * 34];       // 64 d x 32 keys (pad 34)
  __shared__ float          lds_s[4][16 * 33];     // per-wave raw scores 16x32
  __shared__ unsigned short lds_p[4][16 * 34];     // per-wave probs bf16 16x32

  const int tid  = threadIdx.x;
  const int lane = tid & 31;
  const int wave = tid >> 5;      // 0..3
  const int half = lane >> 4;
  const int l16  = lane & 15;

  const int b = blockIdx.z;
  const int h = blockIdx.y;
  const int qbase = blockIdx.x * 64 + wave * 16;
  const size_t bh = (size_t)b * NH + h;

  const unsigned short* Qp   = Q  + (bh * 1024 + qbase) * HD;
  const unsigned short* QOp  = QO + (bh * 1024 + qbase) * HD;
  const unsigned short* Kp   = K  + bh * 1024 * HD;
  const unsigned short* KOp  = KO + bh * 512  * HD;
  const unsigned short* VTp  = VT  + bh * HD * 1024;
  const unsigned short* VOTp = VOT + bh * HD * 512;

  // Q fragments (A layout), two 32-wide d-chunks, for self and cross queries
  Frag aq[2], aqo[2];
  {
    const unsigned int* Qu  = (const unsigned int*)Qp;
    const unsigned int* QOu = (const unsigned int*)QOp;
#pragma unroll
    for (int ch = 0; ch < 2; ++ch)
#pragma unroll
      for (int v = 0; v < 8; ++v) {
        const int idx = l16 * 32 + ((ch * 32 + KOFF(v, half)) >> 1);
        aq[ch].u[v]  = Qu[idx];
        aqo[ch].u[v] = QOu[idx];
      }
  }

  v8f o0 = {}, o1 = {}, o2 = {}, o3 = {};
  float m_run = -3.0e38f, l_run = 0.0f;
  const float scale = 0.125f;  // 1/sqrt(64)

  for (int blk = 0; blk < 48; ++blk) {
    const bool self = blk < 32;
    const int  kb   = self ? blk * 32 : (blk - 32) * 32;
    const unsigned short* kp  = self ? Kp  : KOp;
    const unsigned short* vp  = self ? VTp : VOTp;
    const int  skv  = self ? 1024 : 512;

    __syncthreads();
#if USE_TDM
    // ---- TDM: DMA K tile (32 keys x 64 d) and V^T tile (64 d x 32 keys) ----
    if (wave == 0) {
      tdm_load_2d((unsigned)(size_t)&lds_kt[0], kp + (size_t)kb * HD,
                  /*tensor_d0=*/HD, /*tensor_d1=*/(unsigned)skv, /*stride0=*/HD,
                  /*tile_d0=*/HD, /*tile_d1=*/32,
                  /*pad_interval(32 dw)=*/4, /*pad_amount(1 dw)=*/0);
      tdm_load_2d((unsigned)(size_t)&lds_vt[0], vp + kb,
                  /*tensor_d0=*/(unsigned)skv, /*tensor_d1=*/HD, /*stride0=*/(unsigned)skv,
                  /*tile_d0=*/32, /*tile_d1=*/HD,
                  /*pad_interval(16 dw)=*/3, /*pad_amount(1 dw)=*/0);
      __builtin_amdgcn_s_wait_tensorcnt(0);
    }
#else
    // ---- manual staging fallback ----
    {
      const int row = tid >> 2, sc = (tid & 3) * 16;
      const uint4* src = (const uint4*)(kp + (size_t)(kb + row) * HD + sc);
      uint4 p0 = src[0], p1 = src[1];
      unsigned int* dst = (unsigned int*)lds_kt + row * 33 + (sc >> 1);
      dst[0]=p0.x; dst[1]=p0.y; dst[2]=p0.z; dst[3]=p0.w;
      dst[4]=p1.x; dst[5]=p1.y; dst[6]=p1.z; dst[7]=p1.w;
    }
    {
      const int row = tid >> 1, sc = (tid & 1) * 16;
      const uint4* src = (const uint4*)(vp + (size_t)row * skv + kb + sc);
      uint4 p0 = src[0], p1 = src[1];
      unsigned int* dst = (unsigned int*)lds_vt + row * 17 + (sc >> 1);
      dst[0]=p0.x; dst[1]=p0.y; dst[2]=p0.z; dst[3]=p0.w;
      dst[4]=p1.x; dst[5]=p1.y; dst[6]=p1.z; dst[7]=p1.w;
    }
#endif
    // prefetch next key block (global_prefetch_b8)
    if (blk + 1 < 48) {
      const bool ns = (blk + 1) < 32;
      const unsigned short* nkp = ns ? Kp : KOp;
      const int nkb = ns ? (blk + 1) * 32 : (blk + 1 - 32) * 32;
      __builtin_prefetch(nkp + (size_t)(nkb + (tid >> 2)) * HD, 0, 1);
    }
    __syncthreads();

    const Frag* aQ = self ? aq : aqo;

    // ---- scores: two 16-key subtiles, K-dim 64 as two WMMAs ----
    v8f s0 = {}, s1 = {};
#pragma unroll
    for (int nt = 0; nt < 2; ++nt) {
      v8f acc = {};
#pragma unroll
      for (int ch = 0; ch < 2; ++ch) {
        Frag bk;
#pragma unroll
        for (int v = 0; v < 8; ++v)
          bk.u[v] = ((const unsigned int*)lds_kt)
                        [(nt * 16 + l16) * 33 + ((ch * 32 + KOFF(v, half)) >> 1)];
        acc = __builtin_amdgcn_wmma_f32_16x16x32_bf16(false, aQ[ch].v, false, bk.v,
                                                      (short)0, acc, false, false);
      }
      if (nt == 0) s0 = acc; else s1 = acc;
    }

    // ---- scale + mask, dump raw scores to LDS (C layout -> row major) ----
    {
      float mv0 = 0.f, mv1 = 0.f;
      if (self) {
        mv0 = mask[(size_t)b * 1024 + kb + l16];
        mv1 = mask[(size_t)b * 1024 + kb + 16 + l16];
      }
#pragma unroll
      for (int r = 0; r < 8; ++r) {
        const int m = r + 8 * half;
        lds_s[wave][m * 33 + l16]      = s0[r] * scale + mv0;
        lds_s[wave][m * 33 + 16 + l16] = s1[r] * scale + mv1;
      }
    }
    __syncthreads();

    // ---- streaming softmax per query row (row = l16; halves duplicate) ----
    float factor;
    {
      const float* sr = &lds_s[wave][l16 * 33];
      float bm = -3.0e38f;
#pragma unroll
      for (int j = 0; j < 32; ++j) bm = fmaxf(bm, sr[j]);
      const float nm = fmaxf(m_run, bm);
      factor = __expf(m_run - nm);
      float sum = 0.f;
      unsigned int* pd = (unsigned int*)&lds_p[wave][l16 * 34];
#pragma unroll
      for (int j = 0; j < 16; ++j) {
        const float p0 = __expf(sr[2 * j]     - nm);
        const float p1 = __expf(sr[2 * j + 1] - nm);
        sum += p0 + p1;
        if (half == 0) pd[j] = f2bf_pack(p0, p1);
      }
      l_run = l_run * factor + sum;
      m_run = nm;
    }
    __syncthreads();

    // ---- rescale accumulators by this wave's per-row factor ----
#pragma unroll
    for (int r = 0; r < 8; ++r) {
      const float fr = lane_bcast(factor, half * 8 + r);
      o0[r] *= fr; o1[r] *= fr; o2[r] *= fr; o3[r] *= fr;
    }

    // ---- P (A layout) from LDS, then O += P @ V over 4 d-subtiles ----
    Frag pa;
#pragma unroll
    for (int v = 0; v < 8; ++v)
      pa.u[v] = ((const unsigned int*)lds_p[wave])[l16 * 17 + (KOFF(v, half) >> 1)];
#pragma unroll
    for (int t = 0; t < 4; ++t) {
      Frag bv;
#pragma unroll
      for (int v = 0; v < 8; ++v)
        bv.u[v] = ((const unsigned int*)lds_vt)
                      [(t * 16 + l16) * 17 + (KOFF(v, half) >> 1)];
      v8f& o = (t == 0) ? o0 : (t == 1) ? o1 : (t == 2) ? o2 : o3;
      o = __builtin_amdgcn_wmma_f32_16x16x32_bf16(false, pa.v, false, bv.v,
                                                  (short)0, o, false, false);
    }
  }

  // ---- finalize: divide by l, write fp32 ctx [b, s, h*64 + d] ----
  const float inv_l = 1.0f / l_run;
#pragma unroll
  for (int r = 0; r < 8; ++r) {
    const float il = lane_bcast(inv_l, half * 8 + r);
    const int q = qbase + r + 8 * half;
    float* op = out + ((size_t)b * 1024 + q) * 1024 + h * HD;
    op[ 0 + l16] = o0[r] * il;
    op[16 + l16] = o1[r] * il;
    op[32 + l16] = o2[r] * il;
    op[48 + l16] = o3[r] * il;
  }
}

// =====================================================================
extern "C" void kernel_launch(void* const* d_in, const int* in_sizes, int n_in,
                              void* d_out, int out_size, void* d_ws, size_t ws_size,
                              hipStream_t stream) {
  (void)in_sizes; (void)n_in; (void)out_size; (void)ws_size;
  const float* hs   = (const float*)d_in[0];   // [8,1024,1024]
  const float* hso  = (const float*)d_in[1];   // [8,512,1024]
  const float* mask = (const float*)d_in[2];   // [8,1,1,1024]
  const float* wq  = (const float*)d_in[3];  const float* bq  = (const float*)d_in[4];
  const float* wk  = (const float*)d_in[5];  const float* bk  = (const float*)d_in[6];
  const float* wv  = (const float*)d_in[7];  const float* bv  = (const float*)d_in[8];
  const float* wqo = (const float*)d_in[9];  const float* bqo = (const float*)d_in[10];
  const float* wko = (const float*)d_in[11]; const float* bko = (const float*)d_in[12];
  const float* wvo = (const float*)d_in[13]; const float* bvo = (const float*)d_in[14];

  // Workspace (bf16): q,k,vT,qo : 8*16*1024*64 each ; ko,voT : 8*16*512*64 each
  unsigned short* ws  = (unsigned short*)d_ws;
  const size_t NBIG = (size_t)8 * NH * 1024 * HD;   // 8,388,608
  const size_t NSML = (size_t)8 * NH * 512  * HD;   // 4,194,304
  unsigned short* q_ws   = ws;
  unsigned short* k_ws   = q_ws  + NBIG;
  unsigned short* vT_ws  = k_ws  + NBIG;
  unsigned short* qo_ws  = vT_ws + NBIG;
  unsigned short* ko_ws  = qo_ws + NBIG;
  unsigned short* voT_ws = ko_ws + NSML;            // total 80 MiB

  dim3 bp(256), gp(16, 8, 8), gpo(8, 8, 8);
  proj_kernel<<<gp,  bp, 0, stream>>>(hs,  wq,  bq,  q_ws,   1024, 0);
  proj_kernel<<<gp,  bp, 0, stream>>>(hs,  wk,  bk,  k_ws,   1024, 0);
  proj_kernel<<<gp,  bp, 0, stream>>>(hs,  wv,  bv,  vT_ws,  1024, 1);
  proj_kernel<<<gp,  bp, 0, stream>>>(hs,  wqo, bqo, qo_ws,  1024, 0);
  proj_kernel<<<gpo, bp, 0, stream>>>(hso, wko, bko, ko_ws,   512, 0);
  proj_kernel<<<gpo, bp, 0, stream>>>(hso, wvo, bvo, voT_ws,  512, 1);

  dim3 ba(128), ga(16, NH, 8);
  attn_kernel<<<ga, ba, 0, stream>>>(q_ws, k_ws, vT_ws, qo_ws, ko_ws, voT_ws,
                                     mask, (float*)d_out);
}
